// AttentionBlock_83743272337846
// MI455X (gfx1250) — matrix operand order repaired and checked by reading, not compiled, not run on hardware
//
#include <hip/hip_runtime.h>
#include <hip/hip_bf16.h>

// ---------------------------------------------------------------------------
// Types matching CDNA5 WMMA operand layouts (wave32)
// ---------------------------------------------------------------------------
typedef _Float16 h8   __attribute__((ext_vector_type(8)));    // 16B chunk
typedef _Float16 v16h __attribute__((ext_vector_type(16)));   // A/B frag f16 16x16x32
typedef float    v2f  __attribute__((ext_vector_type(2)));    // A/B frag f32 16x16x4
typedef float    v8f  __attribute__((ext_vector_type(8)));    // C/D frag 16x16 f32

#define SHUF16(lo, hi) __builtin_shufflevector((lo), (hi), 0,1,2,3,4,5,6,7,8,9,10,11,12,13,14,15)

// Async copy global->LDS via inline asm (ROCm 7.2's builtin has a
// descriptor-style signature; the raw instruction is simpler to emit).
// ISA: VDST = per-lane LDS byte address VGPR, VADDR = 64-bit global address.
#define USE_ASYNC_ASM 1

__device__ __forceinline__ void async_load_b128(const void* gp, void* lp) {
#if USE_ASYNC_ASM
  asm volatile("global_load_async_to_lds_b128 %0, %1, off"
               :
               : "v"((uint32_t)(uintptr_t)lp),
                 "v"((unsigned long long)(uintptr_t)gp)
               : "memory");
#else
  *(h8*)lp = *(const h8*)gp;
#endif
}

__device__ __forceinline__ void async_fence() {
#if USE_ASYNC_ASM
#if __has_builtin(__builtin_amdgcn_s_wait_asynccnt)
  __builtin_amdgcn_s_wait_asynccnt(0);
#else
  asm volatile("s_wait_asynccnt 0x0" ::: "memory");
#endif
#endif
}

// Problem constants
#define BATCH 16
#define CCH   512
#define GRP   32
#define CPG   16          // channels per group
#define HW    4096
#define EPS   1e-5f
#define RSQRT_C 0.044194173824159216f   // 1/sqrt(512)

// ---------------------------------------------------------------------------
// Small utility kernels
// ---------------------------------------------------------------------------
__global__ void cvt_f16_kernel(const float* __restrict__ src, _Float16* __restrict__ dst, int n) {
  int i = blockIdx.x * 256 + threadIdx.x;
  if (i < n) dst[i] = (_Float16)src[i];
}

__global__ void copy_f32_kernel(const float* __restrict__ src, float* __restrict__ dst, int n) {
  int i = blockIdx.x * 256 + threadIdx.x;
  if (i < n) dst[i] = src[i];
}

// ---------------------------------------------------------------------------
// GroupNorm statistics: one workgroup per (b,g) slab of 16*4096 contiguous f32
// ---------------------------------------------------------------------------
__global__ __launch_bounds__(256)
void gn_stats_kernel(const float* __restrict__ x, float* __restrict__ stats) {
  const int bg = blockIdx.x;                 // b*32+g
  const float* p = x + (size_t)bg * (CPG * HW);
  float s1 = 0.f, s2 = 0.f;
  for (int idx = threadIdx.x * 4; idx < CPG * HW; idx += 256 * 4) {
    float4 v = *(const float4*)(p + idx);
    s1 += v.x + v.y + v.z + v.w;
    s2 += v.x * v.x + v.y * v.y + v.z * v.z + v.w * v.w;
  }
#pragma unroll
  for (int m = 16; m >= 1; m >>= 1) {
    s1 += __shfl_xor(s1, m, 32);
    s2 += __shfl_xor(s2, m, 32);
  }
  __shared__ float r1[8], r2[8];
  if ((threadIdx.x & 31) == 0) { r1[threadIdx.x >> 5] = s1; r2[threadIdx.x >> 5] = s2; }
  __syncthreads();
  if (threadIdx.x == 0) {
    float a = 0.f, c = 0.f;
#pragma unroll
    for (int w = 0; w < 8; ++w) { a += r1[w]; c += r2[w]; }
    const float inv_n = 1.f / (float)(CPG * HW);
    float mean = a * inv_n;
    float var  = c * inv_n - mean * mean;
    stats[bg * 2 + 0] = mean;
    stats[bg * 2 + 1] = rsqrtf(var + EPS);
  }
}

// ---------------------------------------------------------------------------
// Tiled GEMM: D[m,n] = sum_k Wh[m,k] * srcB[k,n]  (+ bias, + optional extras)
//   NORM=true : srcB is x; apply (x-mean)*rstd*gn_w+gn_b on load; store f16
//               into qkv[proj][b][ch][i] (M = 1536 = wq|wk|wv stacked)
//   NORM=false: srcB is 'out' (f32); epilogue adds bias + residual out[m,n],
//               stores f32 into final output (M = 512)
// Tile: 128x128, K-step 32, 8 waves (4 x 2), double-buffered LDS, 80B row pad.
// A-tile is moved with GLOBAL_LOAD_ASYNC_TO_LDS_B128 (inline asm).
// ---------------------------------------------------------------------------
template <bool NORM>
__global__ __launch_bounds__(256)
void gemm_kernel(const _Float16* __restrict__ Wh,   // [M,512] row-major f16
                 const float*    __restrict__ bias, // [M]
                 const float*    __restrict__ srcB, // [16][512][4096] f32
                 const float*    __restrict__ stats,// [512][2] mean,rstd (NORM)
                 const float*    __restrict__ gnw,
                 const float*    __restrict__ gnb,
                 _Float16*       __restrict__ out16,// qkv buffer (NORM)
                 float*          __restrict__ outf, // final output (!NORM)
                 const float*    __restrict__ resid)// residual (!NORM)
{
  __shared__ alignas(16) _Float16 lA[2][128 * 40];
  __shared__ alignas(16) _Float16 lB[2][128 * 40];

  const int tid  = threadIdx.x;
  const int lane = tid & 31;
  const int lsub = lane & 15;
  const int lh   = lane >> 4;
  const int wid  = tid >> 5;
  const int wm   = wid & 3;     // 4 waves along M (32 rows each)
  const int wn   = wid >> 2;    // 2 waves along N (64 cols each)

  const int nblk = blockIdx.x;          // 0..511
  const int b    = nblk >> 5;           // batch
  const int i0   = (nblk & 31) * 128;   // spatial tile
  const int m0   = blockIdx.y * 128;    // output-channel tile

  v8f acc[2][4] = {};

  float4 fb[4];

  // A tile: 128 rows x 32 k (f16); 512 16B chunks, 2 per thread, async DMA.
  auto loadA = [&](int s, int k0) {
#pragma unroll
    for (int j = 0; j < 2; ++j) {
      int c = tid + j * 256;
      int row = c >> 2, kc = (c & 3) * 8;
      const _Float16* gp = Wh + (size_t)(m0 + row) * CCH + k0 + kc;
      _Float16* lp = &lA[s][row * 40 + kc];
      async_load_b128(gp, lp);
    }
  };
  auto fetchB = [&](int k0) {
#pragma unroll
    for (int p = 0; p < 4; ++p) {
      int ch = p * 8 + (tid >> 5);
      int n  = (tid & 31) * 4;
      fb[p] = *(const float4*)(srcB + ((size_t)b * CCH + k0 + ch) * HW + i0 + n);
    }
  };
  auto storeB = [&](int s, int k0) {
#pragma unroll
    for (int p = 0; p < 4; ++p) {
      int ch  = p * 8 + (tid >> 5);
      int gch = k0 + ch;
      int n   = (tid & 31) * 4;
      float mu = 0.f, rs = 1.f, gw = 1.f, gb = 0.f;
      if (NORM) {
        int gi = gch >> 4;
        mu = stats[(b * GRP + gi) * 2 + 0];
        rs = stats[(b * GRP + gi) * 2 + 1];
        gw = gnw[gch];
        gb = gnb[gch];
      }
      float4 v = fb[p];
      _Float16* d = &lB[s][ch];
      d[(n + 0) * 40] = (_Float16)((v.x - mu) * rs * gw + gb);
      d[(n + 1) * 40] = (_Float16)((v.y - mu) * rs * gw + gb);
      d[(n + 2) * 40] = (_Float16)((v.z - mu) * rs * gw + gb);
      d[(n + 3) * 40] = (_Float16)((v.w - mu) * rs * gw + gb);
    }
  };

  loadA(0, 0);
  fetchB(0);
  storeB(0, 0);
  async_fence();
  __syncthreads();

  for (int kt = 0; kt < 16; ++kt) {
    const int cur = kt & 1;
    const bool more = (kt < 15);
    const int nk = (kt + 1) * 32;
    if (more) {
      loadA(cur ^ 1, nk);     // async DMA overlaps with the WMMA work below
      fetchB(nk);
    }
    if (kt < 14) {            // prefetch B source two stages ahead into GL2
      const int pk = (kt + 2) * 32;
      __builtin_prefetch(srcB + ((size_t)b * CCH + pk + (tid >> 5)) * HW + i0 + (tid & 31) * 4, 0, 1);
    }

    // --- fragments + WMMA on stage 'cur' ---
    v16h af[2], bf[4];
#pragma unroll
    for (int tm = 0; tm < 2; ++tm) {
      const _Float16* base = &lA[cur][(wm * 32 + tm * 16 + lsub) * 40];
      h8 lo = *(const h8*)(base + lh * 8);        // K = lh*8 .. +7
      h8 hi = *(const h8*)(base + 16 + lh * 8);   // K = 16+lh*8 .. +7
      af[tm] = SHUF16(lo, hi);
    }
#pragma unroll
    for (int tn = 0; tn < 4; ++tn) {
      const _Float16* base = &lB[cur][(wn * 64 + tn * 16 + lsub) * 40];
      h8 lo = *(const h8*)(base + lh * 8);        // K = lh*8 .. +7
      h8 hi = *(const h8*)(base + 16 + lh * 8);   // K = 16+lh*8 .. +7
      bf[tn] = SHUF16(lo, hi);
    }
#pragma unroll
    for (int tm = 0; tm < 2; ++tm)
#pragma unroll
      for (int tn = 0; tn < 4; ++tn)
        acc[tm][tn] = __builtin_amdgcn_wmma_f32_16x16x32_f16(
            false, af[tm], false, bf[tn], (short)0, acc[tm][tn], false, false);

    if (more) storeB(cur ^ 1, nk);
    async_fence();            // own async A-tile writes done before barrier
    __syncthreads();
  }

  // --- epilogue ---
#pragma unroll
  for (int tm = 0; tm < 2; ++tm) {
#pragma unroll
    for (int tn = 0; tn < 4; ++tn) {
      const int mbase = m0 + wm * 32 + tm * 16;
      const int ncol  = i0 + wn * 64 + tn * 16 + lsub;
      v8f a = acc[tm][tn];
#pragma unroll
      for (int r = 0; r < 8; ++r) {
        const int m = mbase + r + 8 * lh;        // D layout: M = r + 8*(lane/16)
        float val = a[r] + bias[m];
        if (NORM) {
          const int proj = m >> 9;               // 0:q 1:k 2:v
          const int ch   = m & 511;
          out16[(size_t)proj * ((size_t)BATCH * CCH * HW) +
                ((size_t)b * CCH + ch) * HW + ncol] = (_Float16)val;
        } else {
          const size_t idx = ((size_t)b * CCH + m) * HW + ncol;
          outf[idx] = val + resid[idx];
        }
      }
    }
  }
}

// ---------------------------------------------------------------------------
// Channel attention per (b,g): S = Q K^T / sqrt(C); softmax over d; out=attn V
// One workgroup (8 waves) per (b,g); waves split the 4096 spatial dim.
// ---------------------------------------------------------------------------
__global__ __launch_bounds__(256)
void attn_kernel(const _Float16* __restrict__ qkv,  // [3][16][512][4096] f16
                 float* __restrict__ outbuf)        // [16][512][4096] f32
{
  __shared__ float lS[8][256];
  __shared__ float lT[256];
  __shared__ float lattn[16 * 17];

  const int bg = blockIdx.x;
  const int b = bg >> 5, g = bg & 31;
  const int tid = threadIdx.x, lane = tid & 31, w = tid >> 5;
  const int lsub = lane & 15, lh = lane >> 4;

  const size_t P = (size_t)BATCH * CCH * HW;
  const _Float16* Q = qkv + 0 * P + ((size_t)b * CCH + g * CPG) * HW;
  const _Float16* K = qkv + 1 * P + ((size_t)b * CCH + g * CPG) * HW;
  const _Float16* V = qkv + 2 * P + ((size_t)b * CCH + g * CPG) * HW;

  // Phase 1: partial S over this wave's spatial slice
  v8f s = {};
  const _Float16* qrow = Q + (size_t)lsub * HW;   // A row c = lane%16
  const _Float16* krow = K + (size_t)lsub * HW;   // B col d = lane%16
  for (int ii = w * 512; ii < w * 512 + 512; ii += 32) {
    h8 alo = *(const h8*)(qrow + ii + lh * 8);
    h8 ahi = *(const h8*)(qrow + ii + 16 + lh * 8);
    v16h A = SHUF16(alo, ahi);
    h8 blo = *(const h8*)(krow + ii + lh * 8);
    h8 bhi = *(const h8*)(krow + ii + 16 + lh * 8);
    v16h B = SHUF16(blo, bhi);
    s = __builtin_amdgcn_wmma_f32_16x16x32_f16(false, A, false, B, (short)0, s, false, false);
  }
#pragma unroll
  for (int r = 0; r < 8; ++r)
    lS[w][(r + 8 * lh) * 16 + lsub] = s[r];
  __syncthreads();

  // Phase 2: total + scale (thread t -> element (t/16, t%16))
  {
    float tot = 0.f;
#pragma unroll
    for (int w2 = 0; w2 < 8; ++w2) tot += lS[w2][tid];
    lT[tid] = tot * RSQRT_C;
  }
  __syncthreads();

  // Phase 3: row-wise softmax over d (16 rows, one thread each)
  if (tid < 16) {
    float mx = -3.4e38f;
#pragma unroll
    for (int d = 0; d < 16; ++d) mx = fmaxf(mx, lT[tid * 16 + d]);
    float e[16];
    float sum = 0.f;
#pragma unroll
    for (int d = 0; d < 16; ++d) { e[d] = __expf(lT[tid * 16 + d] - mx); sum += e[d]; }
    const float inv = 1.f / sum;
#pragma unroll
    for (int d = 0; d < 16; ++d) lattn[tid * 17 + d] = e[d] * inv;
  }
  __syncthreads();

  // Phase 4: out = attn @ V over this wave's spatial slice (K = d = 16)
  float* orow = outbuf + ((size_t)b * CCH + g * CPG) * HW;
  for (int i0 = w * 512; i0 < w * 512 + 512; i0 += 16) {
    v8f o = {};
#pragma unroll
    for (int d0 = 0; d0 < 16; d0 += 4) {
      v2f a, bb;
      a.x  = lattn[lsub * 17 + d0 + 2 * lh + 0];
      a.y  = lattn[lsub * 17 + d0 + 2 * lh + 1];
      bb.x = (float)V[(size_t)(d0 + 2 * lh + 0) * HW + i0 + lsub];
      bb.y = (float)V[(size_t)(d0 + 2 * lh + 1) * HW + i0 + lsub];
      o = __builtin_amdgcn_wmma_f32_16x16x4_f32(false, a, false, bb, (short)0, o, false, false);
    }
#pragma unroll
    for (int r = 0; r < 8; ++r)
      orow[(size_t)(r + 8 * lh) * HW + i0 + lsub] = o[r];
  }
}

// ---------------------------------------------------------------------------
// Launch
// ---------------------------------------------------------------------------
extern "C" void kernel_launch(void* const* d_in, const int* in_sizes, int n_in,
                              void* d_out, int out_size, void* d_ws, size_t ws_size,
                              hipStream_t stream) {
  const float* x   = (const float*)d_in[0];
  const float* gnw = (const float*)d_in[1];
  const float* gnb = (const float*)d_in[2];
  const float* wq  = (const float*)d_in[3];
  const float* bq  = (const float*)d_in[4];
  const float* wk  = (const float*)d_in[5];
  const float* bk  = (const float*)d_in[6];
  const float* wv  = (const float*)d_in[7];
  const float* bv  = (const float*)d_in[8];
  const float* wp  = (const float*)d_in[9];
  const float* bp  = (const float*)d_in[10];
  float* out = (float*)d_out;

  // Workspace layout (all 16B aligned)
  char* ws = (char*)d_ws;
  float*    stats = (float*)(ws);                              // 512*2 f32
  float*    bqkv  = (float*)(ws + 4096);                       // 1536 f32
  _Float16* wqkv  = (_Float16*)(ws + 10240);                   // 1536*512 f16
  _Float16* wp16  = (_Float16*)(ws + 10240 + (size_t)1536 * 512 * 2);
  char* p2 = ws + 10240 + (size_t)1536 * 512 * 2 + (size_t)512 * 512 * 2;
  _Float16* qkv  = (_Float16*)p2;                              // 3*16*512*4096 f16
  float* outb = (float*)(p2 + (size_t)3 * BATCH * CCH * HW * 2); // 16*512*4096 f32

  const int NW = 512 * 512;
  cvt_f16_kernel<<<1024, 256, 0, stream>>>(wq, wqkv, NW);
  cvt_f16_kernel<<<1024, 256, 0, stream>>>(wk, wqkv + NW, NW);
  cvt_f16_kernel<<<1024, 256, 0, stream>>>(wv, wqkv + 2 * NW, NW);
  cvt_f16_kernel<<<1024, 256, 0, stream>>>(wp, wp16, NW);
  copy_f32_kernel<<<2, 256, 0, stream>>>(bq, bqkv, 512);
  copy_f32_kernel<<<2, 256, 0, stream>>>(bk, bqkv + 512, 512);
  copy_f32_kernel<<<2, 256, 0, stream>>>(bv, bqkv + 1024, 512);

  gn_stats_kernel<<<512, 256, 0, stream>>>(x, stats);

  // QKV: M = 1536 (12 m-tiles), N = 16 batches * 32 spatial tiles
  gemm_kernel<true><<<dim3(512, 12), 256, 0, stream>>>(
      wqkv, bqkv, x, stats, gnw, gnb, qkv, nullptr, nullptr);

  attn_kernel<<<512, 256, 0, stream>>>(qkv, outb);

  // Projection + residual: M = 512 (4 m-tiles)
  gemm_kernel<false><<<dim3(512, 4), 256, 0, stream>>>(
      wp16, bp, outb, nullptr, nullptr, nullptr, nullptr, out, outb);
}
